// EnsembleRSSM_76295799046362
// MI455X (gfx1250) — matrix operand order repaired and checked by reading, not compiled, not run on hardware
//
#include <hip/hip_runtime.h>
#include <hip/hip_bf16.h>
#include <cstdint>

// ---------------------------------------------------------------------------
// EnsembleRSSM forward for gfx1250 (MI455X).
// Weights packed once to bf16 WMMA-B tiles (~90MB, L2-resident over T=32 steps).
// Activations packed to bf16 WMMA-A fragments by their producer kernels, so the
// GEMM inner loop is pure b128 loads + v_wmma_f32_16x16x32_bf16 (M=16==B).
// ---------------------------------------------------------------------------

#define E_    5
#define B_    16
#define T_    32
#define A_    8
#define EMB_  1024
#define D_    1024
#define S_    32
#define C_    32
#define U_    640

#define KT_IN   33   /* K=1032 padded to 1056 = 33 k-tiles */
#define KT_GRU  52   /* K=1664 */
#define KT_IO   32   /* K=1024 */
#define KT_IS   20   /* K=640  */
#define KT_OO   64   /* K=2048 */
#define KT_OS   20   /* K=640  */
#define N3D     3072

typedef __attribute__((ext_vector_type(16))) __bf16 v16bf;
typedef __attribute__((ext_vector_type(8)))  float  v8f;

__device__ __forceinline__ __bf16 f2bf(float f) {
  uint32_t u = __builtin_bit_cast(uint32_t, f);
  uint32_t r = u + 0x7FFFu + ((u >> 16) & 1u);   // round-to-nearest-even
  return __builtin_bit_cast(__bf16, (uint16_t)(r >> 16));
}

__device__ __forceinline__ uint32_t hash3(uint32_t a, uint32_t b, uint32_t c) {
  uint64_t x = a;
  x = x * 0x9E3779B97F4A7C15ull + b;
  x ^= x >> 29;
  x = x * 0xBF58476D1CE4E5B9ull + c;
  x ^= x >> 32;
  x *= 0x94D049BB133111EBull;
  x ^= x >> 29;
  return (uint32_t)x;
}

// WMMA A-fragment (16-bit, 16x32) packed storage index.
// GEMM lane l (m=l&15, hi=l>>4) element q holds k = kt*32 + hi*8 + (q>>3)*16 + (q&7).
// Layout: [e][kt][lane][q] contiguous, 512 bf16 (=1KB) per k-tile.
__device__ __forceinline__ size_t afrag(int e, int ktiles, int k, int m) {
  int kt = k >> 5, kl = k & 31;
  int lane = (((kl >> 3) & 1) << 4) + m;
  int q = ((kl >> 4) << 3) | (kl & 7);
  return ((size_t)(e * ktiles + kt) * 32 + lane) * 16 + q;
}

// ---------------------------------------------------------------------------
// Pack fp32 [E][K][N] weight into bf16 WMMA-B tiles (32Kx16N per tile):
// lane (n=lane&15, hi=lane>>4) element q holds K=kt*32+hi*8+(q>>3)*16+(q&7).
// ---------------------------------------------------------------------------
__global__ void pack_w_kernel(const float* __restrict__ src, __bf16* __restrict__ dst,
                              int Kreal, int ktiles, int N) {
  int ntiles = N >> 4;
  size_t total = (size_t)E_ * ntiles * ktiles * 512;
  size_t gid = (size_t)blockIdx.x * blockDim.x + threadIdx.x;
  if (gid >= total) return;
  int within = (int)(gid & 511);
  size_t tile = gid >> 9;
  int kt = (int)(tile % ktiles);
  size_t tmp = tile / ktiles;
  int nt = (int)(tmp % ntiles);
  int e  = (int)(tmp / ntiles);
  int lane = within >> 4, q = within & 15;
  int k = kt * 32 + ((lane >> 4) << 3) + ((q >> 3) << 4) + (q & 7);
  int n = nt * 16 + (lane & 15);
  float v = (k < Kreal) ? src[((size_t)e * Kreal + k) * N + n] : 0.0f;
  dst[gid] = f2bf(v);
}

// ---------------------------------------------------------------------------
// GEMM: Y[e][16][N] = Apacked[e] @ Wpacked[e] + bias[e][N]  (bf16 in, fp32 out)
// grid.x = E * N/16/8 ; block = 128 (4 waves, each wave owns 2 N-tiles).
// Inner loop: 3x b128-pair loads + 2 independent WMMA chains.
// ---------------------------------------------------------------------------
__global__ void gemm_wmma_kernel(const __bf16* __restrict__ Ap, const __bf16* __restrict__ W,
                                 const float* __restrict__ bias, float* __restrict__ Y,
                                 int ktiles, int N) {
  int ntiles = N >> 4;
  int grpN = ntiles >> 3;                 // (4 waves * 2 tiles) per block
  int e   = blockIdx.x / grpN;
  int grp = blockIdx.x % grpN;
  int wave = threadIdx.x >> 5;
  int lane = threadIdx.x & 31;
  int nt0 = (grp * 4 + wave) * 2;

  const __bf16* Arow = Ap + (size_t)e * ktiles * 512 + lane * 16;
  const __bf16* W0 = W + ((size_t)(e * ntiles + nt0) * ktiles) * 512 + lane * 16;
  const __bf16* W1 = W0 + (size_t)ktiles * 512;

  v8f acc0 = {}, acc1 = {};
#pragma unroll 4
  for (int kt = 0; kt < ktiles; ++kt) {
    v16bf a  = *(const v16bf*)(Arow + (size_t)kt * 512);
    v16bf b0 = *(const v16bf*)(W0   + (size_t)kt * 512);
    v16bf b1 = *(const v16bf*)(W1   + (size_t)kt * 512);
    __builtin_prefetch((const void*)(W0 + (size_t)(kt + 4) * 512), 0, 1);
    acc0 = __builtin_amdgcn_wmma_f32_16x16x32_bf16(false, a, false, b0,
                                                   (short)0, acc0, false, false);
    acc1 = __builtin_amdgcn_wmma_f32_16x16x32_bf16(false, a, false, b1,
                                                   (short)0, acc1, false, false);
  }
  // C/D layout: VGPR v, lane -> row = v + hi*8, col = lane&15
  int m = lane & 15, hi = lane >> 4;
  float* Yb = Y + (size_t)e * 16 * N;
  int n0 = nt0 * 16 + m;
  float bv0 = bias[(size_t)e * N + n0];
  float bv1 = bias[(size_t)e * N + n0 + 16];
#pragma unroll
  for (int v = 0; v < 8; ++v) {
    int row = v + hi * 8;
    Yb[(size_t)row * N + n0]      = acc0[v] + bv0;
    Yb[(size_t)row * N + n0 + 16] = acc1[v] + bv1;
  }
}

// ---------------------------------------------------------------------------
// LayerNorm(+scale/offset) then SiLU on [E][16][N]; emit bf16 A-fragments.
// ---------------------------------------------------------------------------
__global__ void ln_silu_kernel(const float* __restrict__ Y, const float* __restrict__ s,
                               const float* __restrict__ o, __bf16* __restrict__ dstp,
                               int dst_ktiles, int kOff, int N) {
  int e = blockIdx.x >> 4, b = blockIdx.x & 15;
  const float* row = Y + ((size_t)e * 16 + b) * N;
  int tid = threadIdx.x;
  float sum = 0.f, sq = 0.f;
  for (int j = tid; j < N; j += blockDim.x) { float v = row[j]; sum += v; sq += v * v; }
  for (int off = 16; off > 0; off >>= 1) {
    sum += __shfl_xor(sum, off, 32);
    sq  += __shfl_xor(sq,  off, 32);
  }
  __shared__ float ls[8], lq[8];
  int wave = tid >> 5;
  if ((tid & 31) == 0) { ls[wave] = sum; lq[wave] = sq; }
  __syncthreads();
  if (tid == 0) {
    float a = 0.f, c2 = 0.f;
    for (int w = 0; w < 8; ++w) { a += ls[w]; c2 += lq[w]; }
    ls[0] = a; lq[0] = c2;
  }
  __syncthreads();
  float mean = ls[0] / N;
  float var  = lq[0] / N - mean * mean;
  float inv  = rsqrtf(var + 1e-6f);
  const float* sp = s + (size_t)e * N;
  const float* op = o + (size_t)e * N;
  for (int j = tid; j < N; j += blockDim.x) {
    float y = (row[j] - mean) * inv * sp[j] + op[j];
    float si = y / (1.f + expf(-y));
    dstp[afrag(e, dst_ktiles, kOff + j, b)] = f2bf(si);
  }
}

// ---------------------------------------------------------------------------
// Blend carried state with initial state by is_first; build packed GEMM inputs:
// xin_p[k<1024]=stoch, [1024..1031]=a2, [1032..1055]=0 ; xg_p[k<1024]=dtr.
// ---------------------------------------------------------------------------
__global__ void prepare_x_kernel(float* dtr, float* stc,
                                 const float* __restrict__ idtr, const float* __restrict__ istc,
                                 const float* __restrict__ action, const int* __restrict__ is_first,
                                 __bf16* xin_p, __bf16* xg_p, int t) {
  int e = blockIdx.x >> 4, b = blockIdx.x & 15;
  float f = (float)is_first[b * T_ + t];
  float mm = 1.f - f;
  size_t ro = (size_t)e * 16 + b;
  float* dp = dtr + ro * D_;       const float* ip = idtr + ro * D_;
  float* sp = stc + ro * (S_*C_);  const float* jp = istc + ro * (S_*C_);
  for (int j = threadIdx.x; j < D_; j += blockDim.x) {
    float d = dp[j] * mm + ip[j] * f;
    dp[j] = d;
    xg_p[afrag(e, KT_GRU, j, b)] = f2bf(d);
    float sv = sp[j] * mm + jp[j] * f;
    sp[j] = sv;
    xin_p[afrag(e, KT_IN, j, b)] = f2bf(sv);
  }
  if (threadIdx.x < 32) {                      // k = 1024..1055 (action + pad)
    float v = 0.f;
    if (threadIdx.x < A_) {
      float a = action[((size_t)b * T_ + t) * A_ + threadIdx.x] * mm;
      v = a / fmaxf(1.f, fabsf(a));
    }
    xin_p[afrag(e, KT_IN, S_*C_ + threadIdx.x, b)] = f2bf(v);
  }
}

// GRU gates; new dtr (fp32 carry + packed A), xo_p = [dtr2, embed] packed, outputs.
__global__ void gru_update_kernel(const float* __restrict__ graw, float* dtr,
                                  __bf16* dtr_p, __bf16* xo_p,
                                  const float* __restrict__ embed,
                                  float* __restrict__ out_d1, float* __restrict__ out_d2, int t) {
  int e = blockIdx.x >> 4, b = blockIdx.x & 15;
  size_t ro = (size_t)e * 16 + b;
  const float* g = graw + ro * N3D;
  float* dp = dtr + ro * D_;
  int sel = ((b % E_) == e);
  for (int j = threadIdx.x; j < D_; j += blockDim.x) {
    float r   = 1.f / (1.f + expf(-g[j]));
    float cnd = tanhf(r * g[D_ + j]);
    float u   = 1.f / (1.f + expf(-(g[2 * D_ + j] - 1.f)));
    float d2  = u * cnd + (1.f - u) * dp[j];
    dp[j] = d2;
    __bf16 d2b = f2bf(d2);
    dtr_p[afrag(e, KT_IO, j, b)] = d2b;
    xo_p [afrag(e, KT_OO, j, b)] = d2b;
    xo_p [afrag(e, KT_OO, D_ + j, b)] =
        f2bf(embed[((size_t)b * T_ + t) * EMB_ + j]);
    if (sel) {
      size_t oi = ((size_t)b * T_ + t) * D_ + j;
      out_d1[oi] = d2;
      out_d2[oi] = d2;
    }
  }
}

// softmax over C=32 per wave32 group, unimix, log; gumbel-argmax -> one-hot.
__global__ void unimix_sample_kernel(const float* __restrict__ L,
                                     float* stoch_dst, float* stoch_dst2,
                                     float* out_logit, float* out_stoch,
                                     int t, int salt, int use_gumbel) {
  int e = blockIdx.x >> 4, b = blockIdx.x & 15;
  int tid = threadIdx.x;          // 0..1023 ; wave = stoch group s, lane = class c
  int c = tid & 31;
  float l = L[((size_t)e * 16 + b) * (S_*C_) + tid];
  float mx = l;
  for (int off = 16; off > 0; off >>= 1) mx = fmaxf(mx, __shfl_xor(mx, off, 32));
  float ex = expf(l - mx);
  float sm = ex;
  for (int off = 16; off > 0; off >>= 1) sm += __shfl_xor(sm, off, 32);
  float p  = 0.99f * (ex / sm) + 0.01f / 32.f;
  float lg = logf(p);
  int sel = ((b % E_) == e);
  if (out_logit && sel) out_logit[((size_t)b * T_ + t) * (S_*C_) + tid] = lg;
  float key = lg;
  if (use_gumbel) {
    uint32_t h = hash3(0x9E3779B9u * (uint32_t)salt + (uint32_t)t,
                       (uint32_t)(e * 16 + b), (uint32_t)tid);
    float u = (float)(h >> 8) * (1.0f / 16777216.0f) + 1e-12f;
    key = lg - logf(-logf(u));
  }
  float bestv = key; int besti = c;
  for (int off = 16; off > 0; off >>= 1) {
    float ov = __shfl_xor(bestv, off, 32);
    int   oi = __shfl_xor(besti, off, 32);
    if (ov > bestv || (ov == bestv && oi < besti)) { bestv = ov; besti = oi; }
  }
  float oh = (c == besti) ? 1.f : 0.f;
  size_t si = ((size_t)e * 16 + b) * (S_*C_) + tid;
  if (stoch_dst)  stoch_dst[si]  = oh;
  if (stoch_dst2) stoch_dst2[si] = oh;
  if (out_stoch && sel) out_stoch[((size_t)b * T_ + t) * (S_*C_) + tid] = oh;
}

__global__ void init_state_kernel(const float* __restrict__ p_init, float* dtr,
                                  float* idtr, __bf16* dtr_p) {
  int e = blockIdx.x >> 4, b = blockIdx.x & 15;
  size_t ro = (size_t)e * 16 + b;
  for (int j = threadIdx.x; j < D_; j += blockDim.x) {
    float v = tanhf(p_init[(size_t)e * D_ + j]);
    dtr[ro * D_ + j]  = v;
    idtr[ro * D_ + j] = v;
    dtr_p[afrag(e, KT_IO, j, b)] = f2bf(v);
  }
}

// ---------------------------------------------------------------------------
extern "C" void kernel_launch(void* const* d_in, const int* in_sizes, int n_in,
                              void* d_out, int out_size, void* d_ws, size_t ws_size,
                              hipStream_t stream) {
  (void)in_sizes; (void)n_in;
  const float* embed  = (const float*)d_in[0];
  const float* action = (const float*)d_in[1];
  const int*   isf    = (const int*)  d_in[2];
  const float* p_init = (const float*)d_in[3];
  const float* w_in   = (const float*)d_in[4];
  const float* b_in   = (const float*)d_in[5];
  const float* s_in   = (const float*)d_in[6];
  const float* o_in   = (const float*)d_in[7];
  const float* w_gru  = (const float*)d_in[8];
  const float* b_gru  = (const float*)d_in[9];
  const float* w_io   = (const float*)d_in[10];
  const float* b_io   = (const float*)d_in[11];
  const float* s_io   = (const float*)d_in[12];
  const float* o_io   = (const float*)d_in[13];
  const float* w_is   = (const float*)d_in[14];
  const float* b_is   = (const float*)d_in[15];
  const float* w_oo   = (const float*)d_in[16];
  const float* b_oo   = (const float*)d_in[17];
  const float* s_oo   = (const float*)d_in[18];
  const float* o_oo   = (const float*)d_in[19];
  const float* w_os   = (const float*)d_in[20];
  const float* b_os   = (const float*)d_in[21];
  float* out = (float*)d_out;
  (void)out_size;

  // output sections (6 tensors of B*T*1024, return order of the reference)
  const size_t OSZ = (size_t)B_ * T_ * 1024;
  float* o_post_st = out + 0 * OSZ;
  float* o_deter1  = out + 1 * OSZ;
  float* o_post_lg = out + 2 * OSZ;
  float* o_prio_st = out + 3 * OSZ;
  float* o_deter2  = out + 4 * OSZ;
  float* o_prio_lg = out + 5 * OSZ;

  // ---- deterministic workspace carve-out ----
  size_t off = 0;
  auto take = [&](size_t bytes) -> char* {
    off = (off + 255) & ~(size_t)255;
    char* p = (char*)d_ws + off;
    off += bytes;
    return p;
  };
  // packed bf16 weights (tile = 512 bf16 = 1KB)
  const size_t PW_IN  = (size_t)E_ * (U_/16)   * KT_IN  * 512;
  const size_t PW_GRU = (size_t)E_ * (N3D/16)  * KT_GRU * 512;
  const size_t PW_IO  = (size_t)E_ * (U_/16)   * KT_IO  * 512;
  const size_t PW_IS  = (size_t)E_ * (1024/16) * KT_IS  * 512;
  const size_t PW_OO  = (size_t)E_ * (U_/16)   * KT_OO  * 512;
  const size_t PW_OS  = (size_t)E_ * (1024/16) * KT_OS  * 512;
  __bf16* pw_in  = (__bf16*)take(PW_IN  * 2);
  __bf16* pw_gru = (__bf16*)take(PW_GRU * 2);
  __bf16* pw_io  = (__bf16*)take(PW_IO  * 2);
  __bf16* pw_is  = (__bf16*)take(PW_IS  * 2);
  __bf16* pw_oo  = (__bf16*)take(PW_OO  * 2);
  __bf16* pw_os  = (__bf16*)take(PW_OS  * 2);
  // packed bf16 A-fragment activation buffers
  __bf16* xin_p = (__bf16*)take((size_t)E_ * KT_IN  * 512 * 2);
  __bf16* xg_p  = (__bf16*)take((size_t)E_ * KT_GRU * 512 * 2);
  __bf16* dtr_p = (__bf16*)take((size_t)E_ * KT_IO  * 512 * 2);
  __bf16* h_p   = (__bf16*)take((size_t)E_ * KT_IS  * 512 * 2);
  __bf16* xo_p  = (__bf16*)take((size_t)E_ * KT_OO  * 512 * 2);
  __bf16* ho_p  = (__bf16*)take((size_t)E_ * KT_OS  * 512 * 2);
  // fp32 state / GEMM-output buffers
  const size_t EB = (size_t)E_ * B_;
  float* dtr  = (float*)take(EB * D_    * 4);
  float* stc  = (float*)take(EB * S_*C_ * 4);
  float* idtr = (float*)take(EB * D_    * 4);
  float* istc = (float*)take(EB * S_*C_ * 4);
  float* yraw = (float*)take(EB * U_    * 4);
  float* graw = (float*)take(EB * N3D   * 4);
  float* lraw = (float*)take(EB * 1024  * 4);
  if (off > ws_size) return;   // need ~93MB of scratch

  auto packGrid = [](size_t elems) { return (unsigned)((elems + 255) / 256); };

  // ---- one-time: pack all weights to bf16 WMMA-B tiles (L2-resident after) ----
  pack_w_kernel<<<packGrid(PW_IN),  256, 0, stream>>>(w_in,  pw_in,  1032, KT_IN,  U_);
  pack_w_kernel<<<packGrid(PW_GRU), 256, 0, stream>>>(w_gru, pw_gru, 1664, KT_GRU, N3D);
  pack_w_kernel<<<packGrid(PW_IO),  256, 0, stream>>>(w_io,  pw_io,  1024, KT_IO,  U_);
  pack_w_kernel<<<packGrid(PW_IS),  256, 0, stream>>>(w_is,  pw_is,  640,  KT_IS,  1024);
  pack_w_kernel<<<packGrid(PW_OO),  256, 0, stream>>>(w_oo,  pw_oo,  2048, KT_OO,  U_);
  pack_w_kernel<<<packGrid(PW_OS),  256, 0, stream>>>(w_os,  pw_os,  640,  KT_OS,  1024);

  const unsigned G_EB = (unsigned)EB;           // 80 row-blocks
  const unsigned G640 = E_ * (U_/16)   / 8;     // 25
  const unsigned G1K  = E_ * (1024/16) / 8;     // 40
  const unsigned G3K  = E_ * (N3D/16)  / 8;     // 120

  // ---- initial state: deter=tanh(p_init); stoch=onehot(argmax(unimix)) ----
  init_state_kernel<<<G_EB, 256, 0, stream>>>(p_init, dtr, idtr, dtr_p);
  gemm_wmma_kernel<<<G640, 128, 0, stream>>>(dtr_p, pw_io, b_io, yraw, KT_IO, U_);
  ln_silu_kernel<<<G_EB, 256, 0, stream>>>(yraw, s_io, o_io, h_p, KT_IS, 0, U_);
  gemm_wmma_kernel<<<G1K, 128, 0, stream>>>(h_p, pw_is, b_is, lraw, KT_IS, 1024);
  unimix_sample_kernel<<<G_EB, 1024, 0, stream>>>(lraw, istc, stc, nullptr, nullptr, 0, 0, 0);

  // ---- sequential scan over T ----
  for (int t = 0; t < T_; ++t) {
    prepare_x_kernel<<<G_EB, 256, 0, stream>>>(dtr, stc, idtr, istc, action, isf,
                                               xin_p, xg_p, t);
    gemm_wmma_kernel<<<G640, 128, 0, stream>>>(xin_p, pw_in, b_in, yraw, KT_IN, U_);
    ln_silu_kernel<<<G_EB, 256, 0, stream>>>(yraw, s_in, o_in, xg_p, KT_GRU, D_, U_);
    gemm_wmma_kernel<<<G3K, 128, 0, stream>>>(xg_p, pw_gru, b_gru, graw, KT_GRU, N3D);
    gru_update_kernel<<<G_EB, 256, 0, stream>>>(graw, dtr, dtr_p, xo_p, embed,
                                                o_deter1, o_deter2, t);
    gemm_wmma_kernel<<<G640, 128, 0, stream>>>(dtr_p, pw_io, b_io, yraw, KT_IO, U_);
    ln_silu_kernel<<<G_EB, 256, 0, stream>>>(yraw, s_io, o_io, h_p, KT_IS, 0, U_);
    gemm_wmma_kernel<<<G1K, 128, 0, stream>>>(h_p, pw_is, b_is, lraw, KT_IS, 1024);
    unimix_sample_kernel<<<G_EB, 1024, 0, stream>>>(lraw, nullptr, nullptr,
                                                    o_prio_lg, o_prio_st, t, 1, 1);
    gemm_wmma_kernel<<<G640, 128, 0, stream>>>(xo_p, pw_oo, b_oo, yraw, KT_OO, U_);
    ln_silu_kernel<<<G_EB, 256, 0, stream>>>(yraw, s_oo, o_oo, ho_p, KT_OS, 0, U_);
    gemm_wmma_kernel<<<G1K, 128, 0, stream>>>(ho_p, pw_os, b_os, lraw, KT_OS, 1024);
    unimix_sample_kernel<<<G_EB, 1024, 0, stream>>>(lraw, stc, nullptr,
                                                    o_post_lg, o_post_st, t, 2, 1);
  }
}
